// RegionProposalNetwork_85306640433490
// MI455X (gfx1250) — matrix operand order repaired and checked by reading, not compiled, not run on hardware
//
#include <hip/hip_runtime.h>

#define N_IMG   4
#define CIN     1024
#define CMID    512
#define HH      64
#define WW      64
#define HW      4096
#define NANCH   9
#define NPA     (HW * NANCH)   /* 36864 anchors per image */
#define PRE_NMS 2000
#define POST_NMS 300

typedef __attribute__((ext_vector_type(16))) __bf16 v16bf;
typedef __attribute__((ext_vector_type(8)))  float  v8f;
typedef int v4i_ __attribute__((vector_size(16)));

union Frag { v16bf v; uint4 q[2]; };

__device__ __forceinline__ unsigned short f2bf(float f) {
  unsigned int u = __float_as_uint(f);
  u += 0x7fffu + ((u >> 16) & 1u);        // round-to-nearest-even
  return (unsigned short)(u >> 16);
}
__device__ __forceinline__ float bf2f(unsigned short s) {
  return __uint_as_float(((unsigned int)s) << 16);
}

// CDNA5 async global->LDS copy (ASYNCcnt path), guarded so absence of the
// builtin in a given toolchain falls back to plain load/store.
// Signature (from compiler diagnostic): (v4i AS1*, v4i AS3*, imm offset, imm cpol)
#if defined(__gfx1250__) && __has_builtin(__builtin_amdgcn_global_load_async_to_lds_b128) && __has_builtin(__builtin_amdgcn_s_wait_asynccnt)
#define USE_ASYNC_LDS 1
#else
#define USE_ASYNC_LDS 0
#endif

#define AS1V4(p) ((__attribute__((address_space(1))) v4i_*)(uintptr_t)(p))
#define AS3V4(p) ((__attribute__((address_space(3))) v4i_*)(unsigned int)(uintptr_t)(p))

// ---------------------------------------------------------------------------
// 1) fp32 -> bf16 conversion; weights transposed OIHW -> [k9][oc][ic]
// ---------------------------------------------------------------------------
__global__ void convert_kernel(const float* __restrict__ x,
                               const float* __restrict__ w,
                               unsigned short* __restrict__ xb,
                               unsigned short* __restrict__ wt) {
  long tid = (long)blockIdx.x * blockDim.x + threadIdx.x;
  long stride = (long)gridDim.x * blockDim.x;
  const long nx = (long)N_IMG * CIN * HW;
  for (long i = tid; i < nx; i += stride) xb[i] = f2bf(x[i]);
  const long nw = (long)CMID * CIN * 9;
  for (long i = tid; i < nw; i += stride) {
    int k   = (int)(i % 9);
    long r  = i / 9;
    int ic  = (int)(r % CIN);
    int oc  = (int)(r / CIN);
    wt[((long)k * CMID + oc) * CIN + ic] = f2bf(w[i]);
  }
}

// ---------------------------------------------------------------------------
// 2) conv1 3x3 (1024->512) implicit GEMM, v_wmma_f32_16x16x32_bf16.
//    WG tile: 128 oc x 64 px (one image row); 8 waves, each 32x32 (2x2 WMMA).
//    Loop: outer ic-slab (32 ch) stages one padded activation block
//    [66 cols][3 rows][32 ic] once; inner 9 taps only restage weights (async).
// ---------------------------------------------------------------------------
#define BRS 48   /* halfs per (col,row) cell; 96B => 16B-aligned b128 frags */

__global__ __launch_bounds__(256) void conv1_wmma_kernel(
    const unsigned short* __restrict__ xb,   // [n][ic][h][w] bf16
    const unsigned short* __restrict__ wt,   // [k9][oc][ic] bf16
    const float* __restrict__ bias,          // [512]
    unsigned short* __restrict__ mid) {      // [n][oc][h][w] bf16 (post-ReLU)
  __shared__ __align__(16) unsigned short As[128 * 40];     // [oc][ic32(+pad)]
  __shared__ __align__(16) unsigned short Br[66 * 3 * BRS]; // [col][row][ic32]

  const int tid    = threadIdx.x;
  const int lane   = tid & 31;
  const int wv     = tid >> 5;
  const int waveOc = (wv & 3) * 32;
  const int wavePx = (wv >> 2) * 32;
  const int half   = lane >> 4;
  const int lm     = lane & 15;
  const int hRow   = blockIdx.x;        // output row; px tile = full row of 64
  const int ocBase = blockIdx.y * 128;
  const int n      = blockIdx.z;

  v8f acc[2][2] = {};

  for (int slab = 0; slab < 32; ++slab) {
    const int icBase = slab * 32;

    // ---- stage activation block once per slab: 66x3x32, b128 interior ----
    for (int c = tid; c < 768; c += 256) {      // 96 rowsets x 8 chunks
      int rowset = c >> 3;
      int chunk  = c & 7;                        // cols 1+8*chunk .. +8
      int ic  = rowset / 3;
      int row = rowset - ic * 3;
      int hsrc = hRow + row - 1;
      uint4 qv = {0u, 0u, 0u, 0u};
      if (hsrc >= 0 && hsrc < HH)
        qv = *(const uint4*)(xb +
            (((long)n * CIN + icBase + ic) * HH + hsrc) * WW + chunk * 8);
      const unsigned short* pv = (const unsigned short*)&qv;
      #pragma unroll
      for (int e = 0; e < 8; ++e) {
        int col = chunk * 8 + e + 1;             // wsrc = col-1
        Br[(col * 3 + row) * BRS + ic] = pv[e];
      }
    }
    // zero-pad edge columns (wsrc = -1 and 64)
    for (int c = tid; c < 192; c += 256) {       // 6 cells x 32 ic
      int cell = c >> 5;
      int ic   = c & 31;
      int col  = (cell >= 3) ? 65 : 0;
      int row  = cell % 3;
      Br[(col * 3 + row) * BRS + ic] = 0;
    }

    for (int tap = 0; tap < 9; ++tap) {
      const int ky = tap / 3, kx = tap % 3;

      // ---- stage A: 128 oc x 32 ic weights (async global->LDS if avail) ----
      #pragma unroll
      for (int cc = 0; cc < 2; ++cc) {
        int c    = tid + cc * 256;               // 512 chunks of 8 bf16
        int ocl  = c >> 2;
        int part = c & 3;
        const unsigned short* gsrc = wt +
            ((long)(tap * CMID + ocBase + ocl) * CIN + icBase + part * 8);
        unsigned short* ldst = &As[ocl * 40 + part * 8];
#if USE_ASYNC_LDS
        __builtin_amdgcn_global_load_async_to_lds_b128(AS1V4(gsrc), AS3V4(ldst), 0, 0);
#else
        *(uint4*)ldst = *(const uint4*)gsrc;
#endif
      }
      if (tap + 1 < 9)
        __builtin_prefetch(wt + ((long)((tap + 1) * CMID + ocBase + (tid >> 2)) * CIN +
                                 icBase + (tid & 3) * 8), 0, 1);
#if USE_ASYNC_LDS
      __builtin_amdgcn_s_wait_asynccnt(0);
#endif
      __syncthreads();

      // ---- fragments (ds_load_b128 x2 each) ----
      Frag af[2], bfr[2];
      {
        const int r0 = (waveOc + lm) * 40;
        const int r1 = (waveOc + 16 + lm) * 40;
        af[0].q[0] = *(const uint4*)(&As[r0 + half * 16]);
        af[0].q[1] = *(const uint4*)(&As[r0 + half * 16 + 8]);
        af[1].q[0] = *(const uint4*)(&As[r1 + half * 16]);
        af[1].q[1] = *(const uint4*)(&As[r1 + half * 16 + 8]);
        const int c0 = wavePx + lm + kx;         // Br col = px + kx
        const int c1 = wavePx + 16 + lm + kx;
        const uint4* bp0 = (const uint4*)(&Br[(c0 * 3 + ky) * BRS + half * 16]);
        bfr[0].q[0] = bp0[0]; bfr[0].q[1] = bp0[1];
        const uint4* bp1 = (const uint4*)(&Br[(c1 * 3 + ky) * BRS + half * 16]);
        bfr[1].q[0] = bp1[0]; bfr[1].q[1] = bp1[1];
      }
      #pragma unroll
      for (int mT = 0; mT < 2; ++mT)
        #pragma unroll
        for (int nT = 0; nT < 2; ++nT)
          acc[mT][nT] = __builtin_amdgcn_wmma_f32_16x16x32_bf16(
              false, af[mT].v, false, bfr[nT].v, (short)0, acc[mT][nT],
              false, false);
      __syncthreads();
    }
  }

  // ---- epilogue: bias + ReLU -> bf16 mid ----
  #pragma unroll
  for (int mT = 0; mT < 2; ++mT) {
    #pragma unroll
    for (int nT = 0; nT < 2; ++nT) {
      v8f a = acc[mT][nT];
      int px = wavePx + nT * 16 + lm;
      #pragma unroll
      for (int r = 0; r < 8; ++r) {
        int oc = ocBase + waveOc + mT * 16 + r + 8 * half;
        float v = a[r] + bias[oc];
        v = v > 0.f ? v : 0.f;
        mid[((long)(n * CMID + oc)) * HW + hRow * WW + px] = f2bf(v);
      }
    }
  }
  __builtin_amdgcn_s_wait_tensorcnt(0);
}

// ---------------------------------------------------------------------------
// 3) 1x1 convs (loc: 36ch, score: 18ch) written in transposed/reshaped order.
// ---------------------------------------------------------------------------
__global__ void heads_kernel(const unsigned short* __restrict__ mid,
                             const float* __restrict__ loc_w,
                             const float* __restrict__ loc_b,
                             const float* __restrict__ score_w,
                             const float* __restrict__ score_b,
                             float* __restrict__ out_locs,
                             float* __restrict__ out_scores) {
  long idx = (long)blockIdx.x * blockDim.x + threadIdx.x;
  const long total = (long)N_IMG * 54 * HW;
  if (idx >= total) return;
  int pix = (int)(idx % HW);
  long rest = idx / HW;
  int oc = (int)(rest % 54);
  int n  = (int)(rest / 54);

  const float* wrow;
  float sum;
  if (oc < 36) { wrow = loc_w + oc * CMID;          sum = loc_b[oc]; }
  else         { wrow = score_w + (oc - 36) * CMID; sum = score_b[oc - 36]; }

  const unsigned short* mp = mid + ((long)n * CMID) * HW + pix;
  for (int c = 0; c < CMID; ++c) sum += bf2f(mp[(long)c * HW]) * wrow[c];

  if (oc < 36) {
    int a = oc / 4, comp = oc % 4;
    out_locs[((long)n * NPA + (long)pix * 9 + a) * 4 + comp] = sum;
  } else {
    int s = oc - 36, a = s / 2, comp = s % 2;
    out_scores[((long)n * NPA + (long)pix * 9 + a) * 2 + comp] = sum;
  }
}

// ---------------------------------------------------------------------------
// 4) softmax over (bg,fg) pairs -> fg probability
// ---------------------------------------------------------------------------
__global__ void fgsoft_kernel(const float* __restrict__ scores,
                              float* __restrict__ fg) {
  long idx = (long)blockIdx.x * blockDim.x + threadIdx.x;
  if (idx >= (long)N_IMG * NPA) return;
  float s0 = scores[idx * 2 + 0];
  float s1 = scores[idx * 2 + 1];
  float m = s0 > s1 ? s0 : s1;
  float e0 = __expf(s0 - m), e1 = __expf(s1 - m);
  fg[idx] = e1 / (e0 + e1);
}

// ---------------------------------------------------------------------------
// 5) anchors + loc2bbox + clip + min-size filter; also writes anchor output
// ---------------------------------------------------------------------------
__global__ void boxes_kernel(const float* __restrict__ locs,
                             const float* __restrict__ fg,
                             const int* __restrict__ img_h,
                             const int* __restrict__ img_w,
                             float* __restrict__ boxes,
                             float* __restrict__ scoresF,
                             float* __restrict__ anchors_out) {
  long idx = (long)blockIdx.x * blockDim.x + threadIdx.x;
  if (idx >= (long)N_IMG * NPA) return;
  int i = (int)(idx % NPA);
  int pix = i / 9, a = i % 9;
  int hIdx = pix / WW, wIdx = pix % WW;
  int ridx = a / 3, sidx = a % 3;

  const float ratios[3] = {0.5f, 1.0f, 2.0f};
  const float scales[3] = {8.0f, 16.0f, 32.0f};
  float ha = 16.0f * scales[sidx] * __fsqrt_rn(ratios[ridx]);
  float wa = 16.0f * scales[sidx] * __fsqrt_rn(1.0f / ratios[ridx]);
  float sy = hIdx * 16.0f, sx = wIdx * 16.0f;
  float ay1 = sy + 8.0f - 0.5f * ha, ax1 = sx + 8.0f - 0.5f * wa;
  float ay2 = sy + 8.0f + 0.5f * ha, ax2 = sx + 8.0f + 0.5f * wa;

  if (idx < NPA) {  // batch 0 covers all anchors once
    anchors_out[(long)i * 4 + 0] = ay1;
    anchors_out[(long)i * 4 + 1] = ax1;
    anchors_out[(long)i * 4 + 2] = ay2;
    anchors_out[(long)i * 4 + 3] = ax2;
  }

  float h = ay2 - ay1, w = ax2 - ax1;
  float cy = ay1 + 0.5f * h, cx = ax1 + 0.5f * w;
  float dy = locs[idx * 4 + 0], dx = locs[idx * 4 + 1];
  float dh = locs[idx * 4 + 2], dw = locs[idx * 4 + 3];
  float ncy = dy * h + cy, ncx = dx * w + cx;
  float nh = __expf(dh) * h, nw = __expf(dw) * w;
  float fh = (float)img_h[0], fw = (float)img_w[0];
  float y1 = fminf(fmaxf(ncy - 0.5f * nh, 0.f), fh);
  float x1 = fminf(fmaxf(ncx - 0.5f * nw, 0.f), fw);
  float y2 = fminf(fmaxf(ncy + 0.5f * nh, 0.f), fh);
  float x2 = fminf(fmaxf(ncx + 0.5f * nw, 0.f), fw);
  boxes[idx * 4 + 0] = y1; boxes[idx * 4 + 1] = x1;
  boxes[idx * 4 + 2] = y2; boxes[idx * 4 + 3] = x2;
  bool valid = ((y2 - y1) >= 16.0f) && ((x2 - x1) >= 16.0f);
  scoresF[idx] = valid ? fg[idx] : -__builtin_inff();
}

// ---------------------------------------------------------------------------
// 6) top-2000 by iterative parallel argmax (one block per image)
// ---------------------------------------------------------------------------
__global__ __launch_bounds__(1024) void topk_kernel(
    const float* __restrict__ boxes, float* __restrict__ scoresF,
    float* __restrict__ topScore, float* __restrict__ topBox) {
  __shared__ float sv[1024];
  __shared__ int   si[1024];
  const int n = blockIdx.x, tid = threadIdx.x;
  float* sc = scoresF + (long)n * NPA;
  const float* bx = boxes + (long)n * NPA * 4;

  for (int t = 0; t < PRE_NMS; ++t) {
    float best = -__builtin_inff(); int bi = -1;
    for (int j = tid; j < NPA; j += 1024) {
      float v = sc[j];
      if (v > best) { best = v; bi = j; }
    }
    sv[tid] = best; si[tid] = bi;
    __syncthreads();
    for (int s = 512; s > 0; s >>= 1) {
      if (tid < s && sv[tid + s] > sv[tid]) { sv[tid] = sv[tid + s]; si[tid] = si[tid + s]; }
      __syncthreads();
    }
    if (tid == 0) {
      int bidx = si[0];
      topScore[(long)n * PRE_NMS + t] = sv[0];
      #pragma unroll
      for (int c = 0; c < 4; ++c)
        topBox[((long)n * PRE_NMS + t) * 4 + c] = (bidx >= 0) ? bx[(long)bidx * 4 + c] : 0.f;
      if (bidx >= 0) sc[bidx] = -__builtin_inff();
    }
    __syncthreads();
  }
}

// ---------------------------------------------------------------------------
// 7) greedy NMS (IoU > 0.7) + compact to 300 rois (one block per image)
// ---------------------------------------------------------------------------
__global__ __launch_bounds__(256) void nms_kernel(
    const float* __restrict__ topScore, const float* __restrict__ topBox,
    float* __restrict__ rois, float* __restrict__ roi_idx) {
  __shared__ float y1s[PRE_NMS], x1s[PRE_NMS], y2s[PRE_NMS], x2s[PRE_NMS], ar[PRE_NMS];
  __shared__ int keep[PRE_NMS];
  const int n = blockIdx.x, tid = threadIdx.x;

  for (int j = tid; j < PRE_NMS; j += 256) {
    float a0 = topBox[((long)n * PRE_NMS + j) * 4 + 0];
    float a1 = topBox[((long)n * PRE_NMS + j) * 4 + 1];
    float a2 = topBox[((long)n * PRE_NMS + j) * 4 + 2];
    float a3 = topBox[((long)n * PRE_NMS + j) * 4 + 3];
    y1s[j] = a0; x1s[j] = a1; y2s[j] = a2; x2s[j] = a3;
    ar[j] = (a2 - a0) * (a3 - a1);
    keep[j] = topScore[(long)n * PRE_NMS + j] > -1e37f ? 1 : 0;
  }
  __syncthreads();

  for (int i = 0; i < PRE_NMS; ++i) {
    if (keep[i]) {
      float iy1 = y1s[i], ix1 = x1s[i], iy2 = y2s[i], ix2 = x2s[i], iar = ar[i];
      for (int j = i + 1 + tid; j < PRE_NMS; j += 256) {
        if (keep[j]) {
          float ty = fmaxf(iy1, y1s[j]);
          float tx = fmaxf(ix1, x1s[j]);
          float by = fminf(iy2, y2s[j]);
          float bx = fminf(ix2, x2s[j]);
          float inter = fmaxf(by - ty, 0.f) * fmaxf(bx - tx, 0.f);
          float iou = inter / (iar + ar[j] - inter);
          if (iou > 0.7f) keep[j] = 0;
        }
      }
    }
    __syncthreads();
  }

  if (tid == 0) {
    int cnt = 0;
    for (int j = 0; j < PRE_NMS && cnt < POST_NMS; ++j) {
      if (keep[j]) {
        #pragma unroll
        for (int c = 0; c < 4; ++c)
          rois[((long)n * POST_NMS + cnt) * 4 + c] =
              (c == 0) ? y1s[j] : (c == 1) ? x1s[j] : (c == 2) ? y2s[j] : x2s[j];
        roi_idx[(long)n * POST_NMS + cnt] = (float)n;
        cnt++;
      }
    }
    for (int s = cnt; s < POST_NMS; ++s) {
      #pragma unroll
      for (int c = 0; c < 4; ++c) rois[((long)n * POST_NMS + s) * 4 + c] = 0.f;
      roi_idx[(long)n * POST_NMS + s] = -1.f;
    }
  }
}

// ---------------------------------------------------------------------------
extern "C" void kernel_launch(void* const* d_in, const int* in_sizes, int n_in,
                              void* d_out, int out_size, void* d_ws, size_t ws_size,
                              hipStream_t stream) {
  const float* x       = (const float*)d_in[0];
  const float* conv1_w = (const float*)d_in[1];
  const float* conv1_b = (const float*)d_in[2];
  const float* score_w = (const float*)d_in[3];
  const float* score_b = (const float*)d_in[4];
  const float* loc_w   = (const float*)d_in[5];
  const float* loc_b   = (const float*)d_in[6];
  const int*   img_h   = (const int*)d_in[7];
  const int*   img_w   = (const int*)d_in[8];

  // workspace layout
  unsigned short* xb  = (unsigned short*)d_ws;               // 16,777,216 bf16
  unsigned short* wt  = xb + (long)N_IMG * CIN * HW;         //  4,718,592 bf16
  unsigned short* mid = wt + (long)CMID * CIN * 9;           //  8,388,608 bf16
  float* fg       = (float*)(mid + (long)N_IMG * CMID * HW); //   147,456 f32
  float* boxes    = fg + (long)N_IMG * NPA;                  //   589,824 f32
  float* scoresF  = boxes + (long)N_IMG * NPA * 4;           //   147,456 f32
  float* topScore = scoresF + (long)N_IMG * NPA;             //     8,000 f32
  float* topBox   = topScore + (long)N_IMG * PRE_NMS;        //    32,000 f32

  // output layout (flat concat in return order)
  float* out        = (float*)d_out;
  float* out_locs   = out;                       // 4*36864*4
  float* out_scores = out + 589824;              // 4*36864*2
  float* out_rois   = out + 589824 + 294912;     // 1200*4
  float* out_ridx   = out_rois + 4800;           // 1200
  float* out_anch   = out_ridx + 1200;           // 36864*4

  hipLaunchKernelGGL(convert_kernel, dim3(8192), dim3(256), 0, stream,
                     x, conv1_w, xb, wt);
  hipLaunchKernelGGL(conv1_wmma_kernel, dim3(64, 4, N_IMG), dim3(256), 0, stream,
                     xb, wt, conv1_b, mid);
  hipLaunchKernelGGL(heads_kernel, dim3(3456), dim3(256), 0, stream,
                     mid, loc_w, loc_b, score_w, score_b, out_locs, out_scores);
  hipLaunchKernelGGL(fgsoft_kernel, dim3(576), dim3(256), 0, stream,
                     out_scores, fg);
  hipLaunchKernelGGL(boxes_kernel, dim3(576), dim3(256), 0, stream,
                     out_locs, fg, img_h, img_w, boxes, scoresF, out_anch);
  hipLaunchKernelGGL(topk_kernel, dim3(N_IMG), dim3(1024), 0, stream,
                     boxes, scoresF, topScore, topBox);
  hipLaunchKernelGGL(nms_kernel, dim3(N_IMG), dim3(256), 0, stream,
                     topScore, topBox, out_rois, out_ridx);
}
